// AR_Back_Step_32847909879896
// MI455X (gfx1250) — compile-verified, matrix operand order checked by reading, not run on hardware
//
#include <hip/hip_runtime.h>
#include <hip/hip_bf16.h>
#include <cstdint>
#include <cstddef>

// ---------------------------------------------------------------------------
// MI455X / gfx1250 persistent autoregressive decoder.
//   * All weights converted fp32 -> fp16 once (54 MB, fully L2-resident).
//   * keys/vals precomputed with full-utilization WMMA GEMM.
//   * 600-step recurrence runs in ONE persistent kernel (8 WGs x 256 thr),
//     grid barriers via device atomics; cell states live in registers.
//   * Every dense GEMV uses v_wmma_f32_16x16x32_f16 (weights = B operand,
//     fully utilized -> instruction stream matches the L2-BW roofline).
// ---------------------------------------------------------------------------

typedef __attribute__((ext_vector_type(16))) _Float16 v16h;
typedef __attribute__((ext_vector_type(8)))  _Float16 v8h;
typedef __attribute__((ext_vector_type(8)))  float    v8f;

#define NWG   8
#define TPB   256
#define NMEL  80
#define NHID  1024
#define NATT  640
#define NTXT  640
#define TRES  600
#define TTXT  200

static_assert(NWG * (TPB / 32) == 64, "64 waves: one 16-row hidden chunk each");

// ---- workspace layout: fp16 region then fp32 region -----------------------
static constexpr size_t H_WHH_A = 0;
static constexpr size_t H_WQ    = H_WHH_A + (size_t)4096 * 1024;
static constexpr size_t H_WIH0  = H_WQ    + (size_t)640  * 1024;
static constexpr size_t H_WHH0  = H_WIH0  + (size_t)4096 * 1664;
static constexpr size_t H_WIH1  = H_WHH0  + (size_t)4096 * 1024;
static constexpr size_t H_WHH1  = H_WIH1  + (size_t)4096 * 1024;
static constexpr size_t H_WD1   = H_WHH1  + (size_t)4096 * 1024;
static constexpr size_t H_WD2   = H_WD1   + (size_t)1024 * 1024;
static constexpr size_t H_WC    = H_WD2   + (size_t)1024 * 1024;
static constexpr size_t H_WK    = H_WC    + (size_t)160  * 1024;
static constexpr size_t H_WV    = H_WK    + (size_t)640  * 640;
static constexpr size_t H_TEXT  = H_WV    + (size_t)640  * 640;
static constexpr size_t H_DECIN = H_TEXT  + (size_t)200  * 640;   // [ha_f16 | ctx_f16]
static constexpr size_t H_H0    = H_DECIN + 1664;
static constexpr size_t H_H1    = H_H0    + 1024;
static constexpr size_t H_D1    = H_H1    + 1024;
static constexpr size_t H_D     = H_D1    + 1024;
static constexpr size_t H_TOTAL = H_D     + 1024;
static constexpr size_t F_BASE_BYTES = ((H_TOTAL * 2 + 255) / 256) * 256;

static constexpr size_t F_KEYS   = 0;
static constexpr size_t F_VALS   = F_KEYS   + (size_t)200 * 640;
static constexpr size_t F_Q      = F_VALS   + (size_t)200 * 640;
static constexpr size_t F_SCORES = F_Q      + 640;
static constexpr size_t F_ATTNW  = F_SCORES + 256;
static constexpr size_t F_HA     = F_ATTNW  + 256;
static constexpr size_t F_CTX    = F_HA     + 1024;
static constexpr size_t F_DECOUT = F_CTX    + 640;
static constexpr size_t F_X      = F_DECOUT + 160;
static constexpr size_t F_BAR    = F_X      + 96;

// ---------------------------------------------------------------------------
__device__ __forceinline__ float sigf(float v) { return 1.0f / (1.0f + __expf(-v)); }

// GEMV via WMMA: accumulates rows [rowBase..rowBase+15] of W(MxK,row-major,f16)
// times x(K,f16) into c (result lands in c[0], lanes 0..15).  K % 32 == 0.
__device__ __forceinline__ void wmma_gemv_acc(v8f& c,
    const _Float16* __restrict__ W, const _Float16* __restrict__ x,
    int K, int rowBase, int lane)
{
  const int m = lane & 15, h = lane >> 4;
  const _Float16* wrow = W + (size_t)(rowBase + m) * (size_t)K + 8 * h;
  const _Float16* xv   = x + 8 * h;
  for (int kb = 0; kb < K; kb += 32) {
    __builtin_prefetch((const void*)(wrow + kb + 512), 0, 0);  // global_prefetch_b8
    v8h xlo = *(const v8h*)(xv + kb);
    v8h xhi = *(const v8h*)(xv + kb + 16);
    v8h blo = *(const v8h*)(wrow + kb);
    v8h bhi = *(const v8h*)(wrow + kb + 16);
    v16h a, b;
#pragma unroll
    for (int e = 0; e < 8; ++e) {
      a[e]     = (m == 0) ? xlo[e] : (_Float16)0.0f;   // activation in A row 0
      a[e + 8] = (m == 0) ? xhi[e] : (_Float16)0.0f;
      b[e]     = blo[e];                                // 16 weight rows in B
      b[e + 8] = bhi[e];
    }
    c = __builtin_amdgcn_wmma_f32_16x16x32_f16(false, a, false, b,
                                               (short)0, c, false, false);
  }
}

__device__ __forceinline__ void grid_sync(unsigned* bar)
{
  __threadfence();
  __syncthreads();
  if (threadIdx.x == 0) {
    unsigned gen = __hip_atomic_load(&bar[1], __ATOMIC_ACQUIRE, __HIP_MEMORY_SCOPE_AGENT);
    unsigned arrived = __hip_atomic_fetch_add(&bar[0], 1u, __ATOMIC_ACQ_REL, __HIP_MEMORY_SCOPE_AGENT);
    if (arrived == NWG - 1) {
      __hip_atomic_store(&bar[0], 0u, __ATOMIC_RELAXED, __HIP_MEMORY_SCOPE_AGENT);
      __hip_atomic_fetch_add(&bar[1], 1u, __ATOMIC_RELEASE, __HIP_MEMORY_SCOPE_AGENT);
    } else {
      while (__hip_atomic_load(&bar[1], __ATOMIC_ACQUIRE, __HIP_MEMORY_SCOPE_AGENT) == gen)
        __builtin_amdgcn_s_sleep(1);
    }
  }
  __syncthreads();
}

__device__ __forceinline__ float block_reduce_sum(float v, float* red, int tid)
{
  red[tid] = v; __syncthreads();
  for (int o = TPB / 2; o > 0; o >>= 1) { if (tid < o) red[tid] += red[tid + o]; __syncthreads(); }
  float r = red[0]; __syncthreads();
  return r;
}
__device__ __forceinline__ float block_reduce_max(float v, float* red, int tid)
{
  red[tid] = v; __syncthreads();
  for (int o = TPB / 2; o > 0; o >>= 1) { if (tid < o) red[tid] = fmaxf(red[tid], red[tid + o]); __syncthreads(); }
  float r = red[0]; __syncthreads();
  return r;
}

// ---------------------------------------------------------------------------
// Kernel 1: fp32 -> fp16 weight conversion + state init
// ---------------------------------------------------------------------------
__device__ __forceinline__ void f2h(const float* s, _Float16* d, int n, int t, int nt)
{ for (int i = t; i < n; i += nt) d[i] = (_Float16)s[i]; }
__device__ __forceinline__ void zh(_Float16* d, int n, int t, int nt)
{ for (int i = t; i < n; i += nt) d[i] = (_Float16)0.0f; }
__device__ __forceinline__ void zf(float* d, int n, int t, int nt)
{ for (int i = t; i < n; i += nt) d[i] = 0.0f; }

__global__ void prep_kernel(const float* __restrict__ text,
                            const float* __restrict__ Whh_a, const float* __restrict__ Wq,
                            const float* __restrict__ Wk,    const float* __restrict__ Wv,
                            const float* __restrict__ Wih0,  const float* __restrict__ Whh0,
                            const float* __restrict__ Wih1,  const float* __restrict__ Whh1,
                            const float* __restrict__ Wd1,   const float* __restrict__ Wd2,
                            const float* __restrict__ Wc,
                            _Float16* __restrict__ hb, float* __restrict__ fb)
{
  const int t = blockIdx.x * blockDim.x + threadIdx.x;
  const int nt = gridDim.x * blockDim.x;
  f2h(Whh_a, hb + H_WHH_A, 4096 * 1024, t, nt);
  f2h(Wq,    hb + H_WQ,    640  * 1024, t, nt);
  f2h(Wih0,  hb + H_WIH0,  4096 * 1664, t, nt);
  f2h(Whh0,  hb + H_WHH0,  4096 * 1024, t, nt);
  f2h(Wih1,  hb + H_WIH1,  4096 * 1024, t, nt);
  f2h(Whh1,  hb + H_WHH1,  4096 * 1024, t, nt);
  f2h(Wd1,   hb + H_WD1,   1024 * 1024, t, nt);
  f2h(Wd2,   hb + H_WD2,   1024 * 1024, t, nt);
  f2h(Wc,    hb + H_WC,    160  * 1024, t, nt);
  f2h(Wk,    hb + H_WK,    640  * 640,  t, nt);
  f2h(Wv,    hb + H_WV,    640  * 640,  t, nt);
  f2h(text,  hb + H_TEXT,  200  * 640,  t, nt);
  zh(hb + H_DECIN, 1664, t, nt);
  zh(hb + H_H0,    1024, t, nt);
  zh(hb + H_H1,    1024, t, nt);
  zf(fb + F_X,  NMEL, t, nt);
  zf(fb + F_HA, NHID, t, nt);
  if (t < 2) ((unsigned*)(fb + F_BAR))[t] = 0u;
}

// ---------------------------------------------------------------------------
// Kernel 2: keys = text @ Wk.T, vals = text @ Wv.T  (full-utilization WMMA GEMM)
// one wave per 16x16 output tile, K = 640 in 20 wmma steps.
// ---------------------------------------------------------------------------
__global__ void kv_kernel(const _Float16* __restrict__ text_h,
                          const _Float16* __restrict__ Wk_h,
                          const _Float16* __restrict__ Wv_h,
                          float* __restrict__ keys, float* __restrict__ vals)
{
  const int gw   = (blockIdx.x * blockDim.x + threadIdx.x) >> 5;  // uniform per wave
  const int lane = threadIdx.x & 31;
  if (gw >= 13 * 40 * 2) return;
  const int mat = gw / (13 * 40);
  const int tt  = gw % (13 * 40);
  const int rb  = (tt / 40) * 16;        // text-row tile (200 rows -> 13 tiles)
  const int cb  = (tt % 40) * 16;        // output-col tile
  const _Float16* W = mat ? Wv_h : Wk_h;
  float* out        = mat ? vals : keys;

  const int m = lane & 15, h = lane >> 4;
  int arow = rb + m; if (arow > TTXT - 1) arow = TTXT - 1;   // clamp tail tile
  const _Float16* ap = text_h + (size_t)arow * NTXT + 8 * h;
  const _Float16* bp = W + (size_t)(cb + m) * NTXT + 8 * h;

  v8f c = {};
  for (int kb = 0; kb < NTXT; kb += 32) {
    v8h alo = *(const v8h*)(ap + kb), ahi = *(const v8h*)(ap + kb + 16);
    v8h blo = *(const v8h*)(bp + kb), bhi = *(const v8h*)(bp + kb + 16);
    v16h a, b;
#pragma unroll
    for (int e = 0; e < 8; ++e) { a[e]=alo[e]; a[e+8]=ahi[e]; b[e]=blo[e]; b[e+8]=bhi[e]; }
    c = __builtin_amdgcn_wmma_f32_16x16x32_f16(false, a, false, b, (short)0, c, false, false);
  }
  const int Nj = lane & 15;
  const int Mb = (lane < 16) ? 0 : 8;
#pragma unroll
  for (int v = 0; v < 8; ++v) {
    int row = rb + Mb + v;
    if (row < TTXT) out[(size_t)row * NATT + cb + Nj] = c[v];
  }
}

// ---------------------------------------------------------------------------
// Kernel 3: persistent 600-step decoder
// ---------------------------------------------------------------------------
__global__ __launch_bounds__(TPB, 1)
void decoder_kernel(const float* __restrict__ residual,
                    const float* __restrict__ Wih_a, const float* __restrict__ b_a,
                    const float* __restrict__ v_attn,
                    const float* __restrict__ b0,  const float* __restrict__ b1,
                    const float* __restrict__ bd1, const float* __restrict__ bd2,
                    const float* __restrict__ bc,
                    const float* __restrict__ Wg,  const float* __restrict__ bg,
                    _Float16* __restrict__ hb, float* __restrict__ fb,
                    float* __restrict__ outs)
{
  const int tid   = threadIdx.x;
  const int lane  = tid & 31;
  const int wave  = tid >> 5;
  const int gwave = blockIdx.x * (TPB / 32) + wave;   // 0..63
  const int e0    = gwave * 16;                       // 16 hidden elems per wave

  const _Float16* Whh_a_h = hb + H_WHH_A;
  const _Float16* Wq_h    = hb + H_WQ;
  const _Float16* Wih0_h  = hb + H_WIH0;
  const _Float16* Whh0_h  = hb + H_WHH0;
  const _Float16* Wih1_h  = hb + H_WIH1;
  const _Float16* Whh1_h  = hb + H_WHH1;
  const _Float16* Wd1_h   = hb + H_WD1;
  const _Float16* Wd2_h   = hb + H_WD2;
  const _Float16* Wc_h    = hb + H_WC;
  _Float16* decin_h = hb + H_DECIN;
  _Float16* h0_h    = hb + H_H0;
  _Float16* h1_h    = hb + H_H1;
  _Float16* d1_h    = hb + H_D1;
  _Float16* d_h     = hb + H_D;

  float* keys   = fb + F_KEYS;
  float* vals   = fb + F_VALS;
  float* qv     = fb + F_Q;
  float* scores = fb + F_SCORES;
  float* attnw  = fb + F_ATTNW;
  float* ha_f   = fb + F_HA;
  float* ctx    = fb + F_CTX;
  float* dout_v = fb + F_DECOUT;
  float* xbuf   = fb + F_X;
  unsigned* bar = (unsigned*)(fb + F_BAR);
  float* gates_out = outs + (size_t)TRES * NMEL;

  __shared__ float red[TPB];

  float ca_r = 0.f, c0_r = 0.f, c1_r = 0.f;   // cell state lives in registers

  for (int s = 0; s < TRES; ++s) {
    // ================= LSTM A: z = Wih_a x + Whh_a ha + b ==================
    {
      v8f zi = {}, zfv = {}, zg = {}, zo = {};
      wmma_gemv_acc(zi,  Whh_a_h, decin_h, NHID, 0 * NHID + e0, lane);
      wmma_gemv_acc(zfv, Whh_a_h, decin_h, NHID, 1 * NHID + e0, lane);
      wmma_gemv_acc(zg,  Whh_a_h, decin_h, NHID, 2 * NHID + e0, lane);
      wmma_gemv_acc(zo,  Whh_a_h, decin_h, NHID, 3 * NHID + e0, lane);
      grid_sync(bar);                         // all reads of old ha done
      if (lane < 16) {
        const int row = e0 + lane;
        float Z[4] = { zi[0], zfv[0], zg[0], zo[0] };
#pragma unroll
        for (int g = 0; g < 4; ++g) {
          const float* wr = Wih_a + (size_t)(g * NHID + row) * NMEL;
          float acc = b_a[g * NHID + row];
          for (int k = 0; k < NMEL; ++k) acc += wr[k] * xbuf[k];   // K=80 fp32
          Z[g] += acc;
        }
        ca_r = sigf(Z[1]) * ca_r + sigf(Z[0]) * tanhf(Z[2]);
        float hh = sigf(Z[3]) * tanhf(ca_r);
        ha_f[row] = hh;
        decin_h[row] = (_Float16)hh;
      }
    }
    grid_sync(bar);

    // ================= attention + gate (workgroup 0) ======================
    if (blockIdx.x == 0) {
      for (int t = wave; t < NATT / 16; t += TPB / 32) {          // q = Wq ha
        v8f c = {};
        wmma_gemv_acc(c, Wq_h, decin_h, NHID, t * 16, lane);
        if (lane < 16) qv[t * 16 + lane] = c[0];
      }
      __syncthreads();
      for (int t = wave; t < TTXT; t += TPB / 32) {               // scores
        float sc = 0.f;
        for (int j = lane; j < NATT; j += 32)
          sc += v_attn[j] * tanhf(qv[j] + keys[(size_t)t * NATT + j]);
        for (int o = 16; o > 0; o >>= 1) sc += __shfl_down(sc, o);
        if (lane == 0) scores[t] = sc;
      }
      __syncthreads();
      float lm = -3.0e38f;                                        // softmax
      for (int t = tid; t < TTXT; t += TPB) lm = fmaxf(lm, scores[t]);
      const float mx = block_reduce_max(lm, red, tid);
      float lsum = 0.f;
      for (int t = tid; t < TTXT; t += TPB) {
        float e = __expf(scores[t] - mx); attnw[t] = e; lsum += e;
      }
      const float inv = 1.0f / block_reduce_sum(lsum, red, tid);
      for (int a = tid; a < NATT; a += TPB) {                     // ctx
        float acc = 0.f;
        for (int t = 0; t < TTXT; ++t) acc += attnw[t] * vals[(size_t)t * NATT + a];
        acc *= inv;
        ctx[a] = acc;
        decin_h[NHID + a] = (_Float16)acc;
      }
      __syncthreads();
      float gp = 0.f;                                             // gate
      for (int j = tid; j < NHID + NATT; j += TPB) {
        float dv = (j < NHID) ? ha_f[j] : ctx[j - NHID];
        gp += Wg[j] * dv;
      }
      float gs = block_reduce_sum(gp, red, tid);
      if (tid == 0) gates_out[s] = sigf(gs + bg[0]);
    }
    grid_sync(bar);

    // ================= LSTM 0: Wih0 [ha;ctx] + Whh0 h0 =====================
    {
      v8f zi = {}, zfv = {}, zg = {}, zo = {};
      wmma_gemv_acc(zi,  Wih0_h, decin_h, NHID + NATT, 0 * NHID + e0, lane);
      wmma_gemv_acc(zfv, Wih0_h, decin_h, NHID + NATT, 1 * NHID + e0, lane);
      wmma_gemv_acc(zg,  Wih0_h, decin_h, NHID + NATT, 2 * NHID + e0, lane);
      wmma_gemv_acc(zo,  Wih0_h, decin_h, NHID + NATT, 3 * NHID + e0, lane);
      wmma_gemv_acc(zi,  Whh0_h, h0_h, NHID, 0 * NHID + e0, lane);
      wmma_gemv_acc(zfv, Whh0_h, h0_h, NHID, 1 * NHID + e0, lane);
      wmma_gemv_acc(zg,  Whh0_h, h0_h, NHID, 2 * NHID + e0, lane);
      wmma_gemv_acc(zo,  Whh0_h, h0_h, NHID, 3 * NHID + e0, lane);
      grid_sync(bar);
      if (lane < 16) {
        const int row = e0 + lane;
        float Zi = zi[0]  + b0[row],          Zf = zfv[0] + b0[NHID + row];
        float Zg = zg[0]  + b0[2*NHID + row], Zo = zo[0]  + b0[3*NHID + row];
        c0_r = sigf(Zf) * c0_r + sigf(Zi) * tanhf(Zg);
        h0_h[row] = (_Float16)(sigf(Zo) * tanhf(c0_r));
      }
    }
    grid_sync(bar);

    // ================= LSTM 1: Wih1 h0 + Whh1 h1 ===========================
    {
      v8f zi = {}, zfv = {}, zg = {}, zo = {};
      wmma_gemv_acc(zi,  Wih1_h, h0_h, NHID, 0 * NHID + e0, lane);
      wmma_gemv_acc(zfv, Wih1_h, h0_h, NHID, 1 * NHID + e0, lane);
      wmma_gemv_acc(zg,  Wih1_h, h0_h, NHID, 2 * NHID + e0, lane);
      wmma_gemv_acc(zo,  Wih1_h, h0_h, NHID, 3 * NHID + e0, lane);
      wmma_gemv_acc(zi,  Whh1_h, h1_h, NHID, 0 * NHID + e0, lane);
      wmma_gemv_acc(zfv, Whh1_h, h1_h, NHID, 1 * NHID + e0, lane);
      wmma_gemv_acc(zg,  Whh1_h, h1_h, NHID, 2 * NHID + e0, lane);
      wmma_gemv_acc(zo,  Whh1_h, h1_h, NHID, 3 * NHID + e0, lane);
      grid_sync(bar);
      if (lane < 16) {
        const int row = e0 + lane;
        float Zi = zi[0]  + b1[row],          Zf = zfv[0] + b1[NHID + row];
        float Zg = zg[0]  + b1[2*NHID + row], Zo = zo[0]  + b1[3*NHID + row];
        c1_r = sigf(Zf) * c1_r + sigf(Zi) * tanhf(Zg);
        h1_h[row] = (_Float16)(sigf(Zo) * tanhf(c1_r));
      }
    }
    grid_sync(bar);

    // ================= head + output (workgroup 0) =========================
    if (blockIdx.x == 0) {
      for (int t = wave; t < NHID / 16; t += TPB / 32) {
        v8f c = {};
        wmma_gemv_acc(c, Wd1_h, h1_h, NHID, t * 16, lane);
        if (lane < 16) d1_h[t * 16 + lane] = (_Float16)tanhf(c[0] + bd1[t * 16 + lane]);
      }
      __syncthreads();
      for (int t = wave; t < NHID / 16; t += TPB / 32) {
        v8f c = {};
        wmma_gemv_acc(c, Wd2_h, d1_h, NHID, t * 16, lane);
        if (lane < 16) d_h[t * 16 + lane] = (_Float16)tanhf(c[0] + bd2[t * 16 + lane]);
      }
      __syncthreads();
      for (int t = wave; t < (2 * NMEL) / 16; t += TPB / 32) {
        v8f c = {};
        wmma_gemv_acc(c, Wc_h, d_h, NHID, t * 16, lane);
        if (lane < 16) dout_v[t * 16 + lane] = c[0] + bc[t * 16 + lane];
      }
      __syncthreads();
      if (tid < NMEL) {
        const int trow = TRES - 1 - s;                 // un-flip the scan
        float lsg = dout_v[tid];
        float bb  = dout_v[NMEL + tid];
        float r   = residual[(size_t)trow * NMEL + tid];
        float o   = (r - bb) * __expf(-lsg);
        outs[(size_t)trow * NMEL + tid] = o;
        xbuf[tid] = o;                                 // next step's LSTM-A input
      }
    }
    grid_sync(bar);
  }
}

// ---------------------------------------------------------------------------
extern "C" void kernel_launch(void* const* d_in, const int* in_sizes, int n_in,
                              void* d_out, int out_size, void* d_ws, size_t ws_size,
                              hipStream_t stream)
{
  const float* residual = (const float*)d_in[0];
  const float* text     = (const float*)d_in[1];
  const float* Wih_a    = (const float*)d_in[2];
  const float* Whh_a    = (const float*)d_in[3];
  const float* b_a      = (const float*)d_in[4];
  const float* Wq       = (const float*)d_in[5];
  const float* Wk       = (const float*)d_in[6];
  const float* Wv       = (const float*)d_in[7];
  const float* v_attn   = (const float*)d_in[8];
  const float* Wih0     = (const float*)d_in[9];
  const float* Whh0     = (const float*)d_in[10];
  const float* b0       = (const float*)d_in[11];
  const float* Wih1     = (const float*)d_in[12];
  const float* Whh1     = (const float*)d_in[13];
  const float* b1       = (const float*)d_in[14];
  const float* Wd1      = (const float*)d_in[15];
  const float* bd1      = (const float*)d_in[16];
  const float* Wd2      = (const float*)d_in[17];
  const float* bd2      = (const float*)d_in[18];
  const float* Wc       = (const float*)d_in[19];
  const float* bc       = (const float*)d_in[20];
  const float* Wg       = (const float*)d_in[21];
  const float* bg       = (const float*)d_in[22];

  _Float16* hb = (_Float16*)d_ws;
  float*    fb = (float*)((char*)d_ws + F_BASE_BYTES);

  prep_kernel<<<256, 256, 0, stream>>>(text, Whh_a, Wq, Wk, Wv, Wih0, Whh0,
                                       Wih1, Whh1, Wd1, Wd2, Wc, hb, fb);
  kv_kernel<<<130, 256, 0, stream>>>(hb + H_TEXT, hb + H_WK, hb + H_WV,
                                     fb + F_KEYS, fb + F_VALS);
  decoder_kernel<<<NWG, TPB, 0, stream>>>(residual, Wih_a, b_a, v_attn, b0, b1,
                                          bd1, bd2, bc, Wg, bg, hb, fb,
                                          (float*)d_out);
}
